// DynamicPTHeadSelection_79207786873069
// MI455X (gfx1250) — matrix operand order repaired and checked by reading, not compiled, not run on hardware
//
#include <hip/hip_runtime.h>

#define B_ 4
#define C_ 64
#define P_ 256
#define D_ 512
#define H_ 8
#define R_ 64

typedef __bf16 bf16;
typedef __attribute__((ext_vector_type(16))) __bf16 v16bf;
typedef __attribute__((ext_vector_type(8)))  __bf16 v8bf;
typedef __attribute__((ext_vector_type(8)))  float  v8f;

union V16u { v16bf v; v8bf h[2]; };

__device__ __forceinline__ v8f wmma_bf16(v16bf a, v16bf b, v8f c) {
  // D = A(16x32) * B(32x16) + C, f32 accumulate
  return __builtin_amdgcn_wmma_f32_16x16x32_bf16(false, a, false, b, (short)0, c, false, false);
}

// A-operand fragment (16x32 MxK) from f32 row-major source (leading dim ld), cvt to bf16.
// Layout: lanes 0-15 hold K[kb..kb+7] & K[kb+16..kb+23]; lanes 16-31 hold K[kb+8..15] & K[kb+24..31].
__device__ __forceinline__ v16bf afrag_f32(const float* base, int ld, int m0, int kb, int lane) {
  const int half = lane >> 4;
  const int m = m0 + (lane & 15);
  const float* p0 = base + (size_t)m * ld + kb + half * 8;
  const float* p1 = p0 + 16;
  v16bf a;
#pragma unroll
  for (int j = 0; j < 8; ++j) { a[j] = (bf16)p0[j]; a[j + 8] = (bf16)p1[j]; }
  return a;
}

// A-operand fragment from bf16 source (LDS or global).
__device__ __forceinline__ v16bf afrag_bf16(const bf16* base, int ld, int m0, int kb, int lane) {
  const int half = lane >> 4;
  const int m = m0 + (lane & 15);
  const bf16* p = base + (size_t)m * ld + kb + half * 8;
  V16u u;
  u.h[0] = *(const v8bf*)p;
  u.h[1] = *(const v8bf*)(p + 16);
  return u.v;
}

// B-operand fragment (32x16 KxN): source stores B^T row-major, i.e. row n holds column n over K.
// Lanes 0-15: K[kb..kb+7] & K[kb+16..kb+23]; lanes 16-31: K[kb+8..15] & K[kb+24..31].
__device__ __forceinline__ v16bf bfrag_bf16(const bf16* base, int ld, int n0, int kb, int lane) {
  const int half = lane >> 4;
  const int n = n0 + (lane & 15);
  const bf16* p = base + (size_t)n * ld + kb + half * 8;
  V16u u;
  u.h[0] = *(const v8bf*)p;
  u.h[1] = *(const v8bf*)(p + 16);
  return u.v;
}

// RoPE: inv_freq[i] = 10000^(-i/32) for i = r & 31
__device__ __forceinline__ void rope_cs(float pos, int r, float* co, float* si) {
  const float invf = __expf(-0.2878231366242558f * (float)(r & 31)); // ln(10000)/32
  __sincosf(pos * invf, si, co);
}

// ---------------------------------------------------------------------------
// Elementwise f32 -> bf16 (n8 = n/8, 16B stores)
__global__ __launch_bounds__(256) void cvt_bf16_kernel(const float* __restrict__ in,
                                                       bf16* __restrict__ out, int n8) {
  const int i = blockIdx.x * 256 + threadIdx.x;
  if (i < n8) {
    const float* p = in + (size_t)i * 8;
    v8bf o;
#pragma unroll
    for (int j = 0; j < 8; ++j) o[j] = (bf16)p[j];
    *(v8bf*)(out + (size_t)i * 8) = o;
  }
}

// ---------------------------------------------------------------------------
// qzT[b,c,d,p] = qz[b,c,p,d]  (bf16) -- B-operand for time AV GEMM
__global__ __launch_bounds__(256) void transpose_t_kernel(const float* __restrict__ qz,
                                                          bf16* __restrict__ qzT) {
  __shared__ bf16 tile[32][33];
  const int bc = blockIdx.x;
  const int p0 = blockIdx.y * 32, d0 = blockIdx.z * 32;
  const float* in = qz + (size_t)bc * (P_ * D_);
  bf16* outp = qzT + (size_t)bc * (D_ * P_);
  const int tx = threadIdx.x & 31, ty = threadIdx.x >> 5;
#pragma unroll
  for (int i = 0; i < 32; i += 8)
    tile[ty + i][tx] = (bf16)in[(size_t)(p0 + ty + i) * D_ + d0 + tx];
  __syncthreads();
#pragma unroll
  for (int i = 0; i < 32; i += 8)
    outp[(size_t)(d0 + ty + i) * P_ + p0 + tx] = tile[tx][ty + i];
}

// qzcT[b,p,d,c] = qz[b,c,p,d]  (bf16) -- B-operand for channel AV GEMM
__global__ __launch_bounds__(256) void transpose_c_kernel(const float* __restrict__ qz,
                                                          bf16* __restrict__ qzcT) {
  __shared__ bf16 tile[32][33];
  const int bp = blockIdx.x;
  const int b = bp >> 8, p = bp & 255;
  const int c0 = blockIdx.y * 32, d0 = blockIdx.z * 32;
  const float* in = qz + (size_t)b * (C_ * P_ * D_) + (size_t)p * D_;
  bf16* outp = qzcT + (size_t)bp * (D_ * C_);
  const int tx = threadIdx.x & 31, ty = threadIdx.x >> 5;
#pragma unroll
  for (int i = 0; i < 32; i += 8)
    tile[ty + i][tx] = (bf16)in[(size_t)(c0 + ty + i) * (P_ * D_) + d0 + tx];
  __syncthreads();
#pragma unroll
  for (int i = 0; i < 32; i += 8)
    outp[(size_t)(d0 + ty + i) * C_ + c0 + tx] = tile[tx][ty + i];
}

// ---------------------------------------------------------------------------
// Time attention scores, fused: per (b,c) WG (256 thr = 8 waves), loop heads:
//   proj (WMMA, bf16 operands from global) -> bias+RoPE in registers -> LDS qu/qv
//   -> logits (WMMA from LDS) -> softmax in LDS -> accumulate Abar/H to global f32.
// Dynamic LDS: 2*256*64*2 + 256*256*4 = 327680 B (full 320 KB WGP allocation).
__global__ __launch_bounds__(256) void time_scores_kernel(
    const bf16* __restrict__ qzb, const bf16* __restrict__ U,
    const bf16* __restrict__ V, const float* __restrict__ bias,
    float* __restrict__ abar) {
  extern __shared__ char smem[];
  bf16* quL = (bf16*)smem;                  // [256][64]
  bf16* qvL = quL + P_ * R_;                // [256][64]
  float* S  = (float*)(qvL + P_ * R_);      // [256][256]

  const int bc = blockIdx.x;
  const int b = bc >> 6;
  const int tid = threadIdx.x;
  const int wave = tid >> 5, lane = tid & 31;
  const int half = lane >> 4, ln = lane & 15;

  const bf16* qzbc = qzb + (size_t)bc * (P_ * D_);   // [256,512] bf16
  const bf16* Ub = U + (size_t)b * (D_ * D_);
  const bf16* Vb = V + (size_t)b * (D_ * D_);
  float* abc = abar + (size_t)bc * (P_ * P_);

  for (int h = 0; h < H_; ++h) {
    // ---- Stage 1: qu/qv = RoPE(qz[b,c] @ W_h^T (+bias))  -> LDS bf16
    for (int mat = 0; mat < 2; ++mat) {
      const bf16* W = (mat == 0 ? Ub : Vb) + (size_t)h * R_ * D_;  // [64,512], row o over k
      bf16* dst = (mat == 0) ? quL : qvL;
      const int m0 = wave * 32;  // 32 p-rows per wave
      v8f acc[2][4] = {};
      for (int kb = 0; kb < D_; kb += 32) {
        v16bf a0 = afrag_bf16(qzbc, D_, m0, kb, lane);
        v16bf a1 = afrag_bf16(qzbc, D_, m0 + 16, kb, lane);
#pragma unroll
        for (int j = 0; j < 4; ++j) {
          v16bf bb = bfrag_bf16(W, D_, j * 16, kb, lane);
          acc[0][j] = wmma_bf16(a0, bb, acc[0][j]);
          acc[1][j] = wmma_bf16(a1, bb, acc[1][j]);
        }
      }
      if (mat == 0) {  // patch_query_bias[b,p,h,r] added before RoPE (qu only)
#pragma unroll
        for (int mi = 0; mi < 2; ++mi)
#pragma unroll
          for (int j = 0; j < 4; ++j)
#pragma unroll
            for (int vj = 0; vj < 8; ++vj) {
              const int p = m0 + mi * 16 + vj + 8 * half;
              const int r = j * 16 + ln;
              acc[mi][j][vj] += bias[(((size_t)b * P_ + p) * H_ + h) * R_ + r];
            }
      }
      // RoPE: partner r^32 lives in accumulator tile j^2, same lane.
#pragma unroll
      for (int mi = 0; mi < 2; ++mi)
#pragma unroll
        for (int j = 0; j < 4; ++j)
#pragma unroll
          for (int vj = 0; vj < 8; ++vj) {
            const int p = m0 + mi * 16 + vj + 8 * half;
            const int r = j * 16 + ln;
            const float x  = acc[mi][j][vj];
            const float xp = acc[mi][j ^ 2][vj];
            float co, si; rope_cs((float)p, r, &co, &si);
            const float rot = (r < 32) ? -xp : xp;
            dst[p * R_ + r] = (bf16)(x * co + rot * si);
          }
    }
    __syncthreads();
    // ---- Stage 2: S = (quL @ qvL^T) / 64
    {
      const int m0 = wave * 32;
      v16bf a[2][2];
#pragma unroll
      for (int mi = 0; mi < 2; ++mi)
#pragma unroll
        for (int ks = 0; ks < 2; ++ks)
          a[mi][ks] = afrag_bf16(quL, R_, m0 + mi * 16, ks * 32, lane);
      for (int jt = 0; jt < 16; ++jt) {
        v8f s0 = {}, s1 = {};
#pragma unroll
        for (int ks = 0; ks < 2; ++ks) {
          v16bf bb = bfrag_bf16(qvL, R_, jt * 16, ks * 32, lane);
          s0 = wmma_bf16(a[0][ks], bb, s0);
          s1 = wmma_bf16(a[1][ks], bb, s1);
        }
#pragma unroll
        for (int vj = 0; vj < 8; ++vj) {
          const int p = m0 + vj + 8 * half;
          const int q = jt * 16 + ln;
          S[p * P_ + q]        = s0[vj] * (1.0f / 64.0f);
          S[(p + 16) * P_ + q] = s1[vj] * (1.0f / 64.0f);
        }
      }
    }
    __syncthreads();
    // ---- row softmax (thread tid owns row tid; rotated scan -> no bank conflicts)
    {
      float* row = S + tid * P_;
      float mx = -3.402823466e38f;
      for (int j = 0; j < P_; ++j) { const int q = (tid + j) & (P_ - 1); mx = fmaxf(mx, row[q]); }
      float sum = 0.f;
      for (int j = 0; j < P_; ++j) { const int q = (tid + j) & (P_ - 1); sum += __expf(row[q] - mx); }
      const float inv = 1.0f / sum;
      for (int j = 0; j < P_; ++j) { const int q = (tid + j) & (P_ - 1); row[q] = __expf(row[q] - mx) * inv; }
    }
    __syncthreads();
    // ---- accumulate head-mean into global Abar (coalesced)
    for (int idx = tid; idx < P_ * P_; idx += 256) {
      const float v = S[idx] * 0.125f;
      abc[idx] = (h == 0) ? v : (abc[idx] + v);
    }
    // next head's S writes are fenced by the stage-1 __syncthreads
  }
}

// ---------------------------------------------------------------------------
// Channel attention scores, fused, per (b,p) WG.
__global__ __launch_bounds__(256) void chan_scores_kernel(
    const bf16* __restrict__ qzb, const bf16* __restrict__ U,
    const bf16* __restrict__ V, float* __restrict__ abar) {
  __shared__ bf16 quL[C_ * R_];
  __shared__ bf16 qvL[C_ * R_];
  __shared__ float S[C_ * C_];
  const int bp = blockIdx.x;
  const int b = bp >> 8, p = bp & 255;
  const int tid = threadIdx.x, wave = tid >> 5, lane = tid & 31;
  const int half = lane >> 4, ln = lane & 15;
  const bf16* qzbp = qzb + (size_t)b * (C_ * P_ * D_) + (size_t)p * D_;  // row c, stride P_*D_
  const bf16* Ub = U + (size_t)b * (D_ * D_);
  const bf16* Vb = V + (size_t)b * (D_ * D_);
  float* abc = abar + (size_t)bp * (C_ * C_);

  for (int h = 0; h < H_; ++h) {
    // Stage 1: waves 0-3 build qu (row tile = wave), waves 4-7 build qv.
    {
      const int mat = wave >> 2;
      const int mt = wave & 3;
      const bf16* W = (mat == 0 ? Ub : Vb) + (size_t)h * R_ * D_;
      bf16* dst = (mat == 0) ? quL : qvL;
      v8f acc[4] = {};
      for (int kb = 0; kb < D_; kb += 32) {
        v16bf a = afrag_bf16(qzbp, P_ * D_, mt * 16, kb, lane);
#pragma unroll
        for (int j = 0; j < 4; ++j) {
          v16bf bb = bfrag_bf16(W, D_, j * 16, kb, lane);
          acc[j] = wmma_bf16(a, bb, acc[j]);
        }
      }
#pragma unroll
      for (int j = 0; j < 4; ++j)
#pragma unroll
        for (int vj = 0; vj < 8; ++vj) {
          const int cch = mt * 16 + vj + 8 * half;
          const int r = j * 16 + ln;
          const float x  = acc[j][vj];
          const float xp = acc[j ^ 2][vj];
          float co, si; rope_cs((float)cch, r, &co, &si);   // RoPE over channel position
          const float rot = (r < 32) ? -xp : xp;
          dst[cch * R_ + r] = (bf16)(x * co + rot * si);
        }
    }
    __syncthreads();
    // Stage 2: S[c,f] = (qu @ qv^T) / 64 ; wave -> (mtile = w&3, jtiles = {(w>>2)*2, +1})
    {
      const int mt = wave & 3;
      const int jb = (wave >> 2) * 2;
      v16bf a[2];
#pragma unroll
      for (int ks = 0; ks < 2; ++ks) a[ks] = afrag_bf16(quL, R_, mt * 16, ks * 32, lane);
#pragma unroll
      for (int jj = 0; jj < 2; ++jj) {
        const int jt = jb + jj;
        v8f s = {};
#pragma unroll
        for (int ks = 0; ks < 2; ++ks) {
          v16bf bb = bfrag_bf16(qvL, R_, jt * 16, ks * 32, lane);
          s = wmma_bf16(a[ks], bb, s);
        }
#pragma unroll
        for (int vj = 0; vj < 8; ++vj) {
          const int cc = mt * 16 + vj + 8 * half;
          S[cc * C_ + jt * 16 + ln] = s[vj] * (1.0f / 64.0f);
        }
      }
    }
    __syncthreads();
    if (tid < C_) {
      float* row = S + tid * C_;
      float mx = -3.402823466e38f;
      for (int j = 0; j < C_; ++j) { const int q = (tid + j) & (C_ - 1); mx = fmaxf(mx, row[q]); }
      float sum = 0.f;
      for (int j = 0; j < C_; ++j) { const int q = (tid + j) & (C_ - 1); sum += __expf(row[q] - mx); }
      const float inv = 1.0f / sum;
      for (int j = 0; j < C_; ++j) { const int q = (tid + j) & (C_ - 1); row[q] = __expf(row[q] - mx) * inv; }
    }
    __syncthreads();
    for (int idx = tid; idx < C_ * C_; idx += 256) {
      const float v = S[idx] * 0.125f;
      abc[idx] = (h == 0) ? v : (abc[idx] + v);
    }
  }
}

// ---------------------------------------------------------------------------
// m_t[b,c] = Abar_t[b,c] @ qz[b,c]   (M=256, N=512, K=256); WG tile 128x128.
__global__ __launch_bounds__(256) void time_av_kernel(
    const float* __restrict__ abar, const bf16* __restrict__ qzT,
    float* __restrict__ out) {
  const int bc = blockIdx.x;
  const int wave = threadIdx.x >> 5;
  const int m0 = blockIdx.y * 128 + (wave >> 1) * 32;
  const int n0 = blockIdx.z * 128 + (wave & 1) * 64;
  const int lane = threadIdx.x & 31, half = lane >> 4, ln = lane & 15;
  const float* A = abar + (size_t)bc * (P_ * P_);
  const bf16* Bm = qzT + (size_t)bc * (D_ * P_);   // row d over q
  float* O = out + (size_t)bc * (P_ * D_);
  v8f acc[2][4] = {};
  for (int kb = 0; kb < P_; kb += 32) {
    if (kb + 32 < P_)  // warm L0/GL2 for next A rows (global_prefetch_b8)
      __builtin_prefetch(A + (size_t)(m0 + (threadIdx.x & 31)) * P_ + kb + 32, 0, 3);
    v16bf a0 = afrag_f32(A, P_, m0, kb, lane);
    v16bf a1 = afrag_f32(A, P_, m0 + 16, kb, lane);
#pragma unroll
    for (int j = 0; j < 4; ++j) {
      v16bf bb = bfrag_bf16(Bm, P_, n0 + j * 16, kb, lane);
      acc[0][j] = wmma_bf16(a0, bb, acc[0][j]);
      acc[1][j] = wmma_bf16(a1, bb, acc[1][j]);
    }
  }
#pragma unroll
  for (int mi = 0; mi < 2; ++mi)
#pragma unroll
    for (int j = 0; j < 4; ++j)
#pragma unroll
      for (int vj = 0; vj < 8; ++vj) {
        const int pp = m0 + mi * 16 + vj + 8 * half;
        const int d = n0 + j * 16 + ln;
        O[(size_t)pp * D_ + d] = acc[mi][j][vj];
      }
}

// m_c[b,:,p,:] = Abar_c[b,p] @ qz[b,:,p,:]  (M=64, N=512, K=64); WG tile 64x128.
__global__ __launch_bounds__(256) void chan_av_kernel(
    const float* __restrict__ abar, const bf16* __restrict__ qzcT,
    float* __restrict__ out) {
  const int bp = blockIdx.x;
  const int b = bp >> 8, p = bp & 255;
  const int wave = threadIdx.x >> 5;
  const int m0 = (wave >> 1) * 16;
  const int n0 = blockIdx.y * 128 + (wave & 1) * 64;
  const int lane = threadIdx.x & 31, half = lane >> 4, ln = lane & 15;
  const float* A = abar + (size_t)bp * (C_ * C_);
  const bf16* Bm = qzcT + (size_t)bp * (D_ * C_);  // row d over f
  float* O = out + (size_t)b * (C_ * P_ * D_) + (size_t)p * D_;
  v8f acc[4] = {};
  for (int kb = 0; kb < C_; kb += 32) {
    v16bf a = afrag_f32(A, C_, m0, kb, lane);
#pragma unroll
    for (int j = 0; j < 4; ++j) {
      v16bf bb = bfrag_bf16(Bm, C_, n0 + j * 16, kb, lane);
      acc[j] = wmma_bf16(a, bb, acc[j]);
    }
  }
#pragma unroll
  for (int j = 0; j < 4; ++j)
#pragma unroll
    for (int vj = 0; vj < 8; ++vj) {
      const int cc = m0 + vj + 8 * half;
      const int d = n0 + j * 16 + ln;
      O[(size_t)cc * (P_ * D_) + d] = acc[j][vj];
    }
}

// ---------------------------------------------------------------------------
extern "C" void kernel_launch(void* const* d_in, const int* in_sizes, int n_in,
                              void* d_out, int out_size, void* d_ws, size_t ws_size,
                              hipStream_t stream) {
  (void)in_sizes; (void)n_in; (void)out_size; (void)ws_size;
  const float* qz   = (const float*)d_in[0];
  const float* tu   = (const float*)d_in[1];
  const float* tv   = (const float*)d_in[2];
  const float* cu   = (const float*)d_in[3];
  const float* cv   = (const float*)d_in[4];
  const float* bias = (const float*)d_in[5];
  float* out_mt = (float*)d_out;
  float* out_mc = out_mt + (size_t)B_ * C_ * P_ * D_;

  char* ws = (char*)d_ws;
  bf16* qzT = (bf16*)ws;        ws += (size_t)B_ * C_ * D_ * P_ * sizeof(bf16);     //  64 MiB
  bf16* qzcT = (bf16*)ws;       ws += (size_t)B_ * P_ * D_ * C_ * sizeof(bf16);     //  64 MiB
  float* abar_t = (float*)ws;   ws += (size_t)B_ * C_ * P_ * P_ * sizeof(float);    //  64 MiB
  float* abar_c = (float*)ws;   ws += (size_t)B_ * P_ * C_ * C_ * sizeof(float);    //  16 MiB
  bf16* qzb = (bf16*)ws;        ws += (size_t)B_ * C_ * P_ * D_ * sizeof(bf16);     //  64 MiB
  bf16* tub = (bf16*)ws;        ws += (size_t)B_ * D_ * D_ * sizeof(bf16);          //   2 MiB
  bf16* tvb = (bf16*)ws;        ws += (size_t)B_ * D_ * D_ * sizeof(bf16);
  bf16* cub = (bf16*)ws;        ws += (size_t)B_ * D_ * D_ * sizeof(bf16);
  bf16* cvb = (bf16*)ws;        ws += (size_t)B_ * D_ * D_ * sizeof(bf16);

  const int tsLds = 2 * P_ * R_ * (int)sizeof(bf16) + P_ * P_ * (int)sizeof(float); // 327680
  (void)hipFuncSetAttribute(reinterpret_cast<const void*>(time_scores_kernel),
                            hipFuncAttributeMaxDynamicSharedMemorySize, tsLds);

  // one-time bf16 conversions (weights + qz) so the hot GEMM loops carry no cvt VALU
  const int nqz8 = (B_ * C_ * P_ * D_) / 8;
  const int nw8  = (B_ * D_ * D_) / 8;
  cvt_bf16_kernel<<<dim3((nqz8 + 255) / 256), 256, 0, stream>>>(qz, qzb, nqz8);
  cvt_bf16_kernel<<<dim3((nw8 + 255) / 256), 256, 0, stream>>>(tu, tub, nw8);
  cvt_bf16_kernel<<<dim3((nw8 + 255) / 256), 256, 0, stream>>>(tv, tvb, nw8);
  cvt_bf16_kernel<<<dim3((nw8 + 255) / 256), 256, 0, stream>>>(cu, cub, nw8);
  cvt_bf16_kernel<<<dim3((nw8 + 255) / 256), 256, 0, stream>>>(cv, cvb, nw8);

  transpose_t_kernel<<<dim3(B_ * C_, P_ / 32, D_ / 32), 256, 0, stream>>>(qz, qzT);
  transpose_c_kernel<<<dim3(B_ * P_, C_ / 32, D_ / 32), 256, 0, stream>>>(qz, qzcT);
  time_scores_kernel<<<dim3(B_ * C_), 256, tsLds, stream>>>(qzb, tub, tvb, bias, abar_t);
  chan_scores_kernel<<<dim3(B_ * P_), 256, 0, stream>>>(qzb, cub, cvb, abar_c);
  time_av_kernel<<<dim3(B_ * C_, P_ / 128, D_ / 128), 256, 0, stream>>>(abar_t, qzT, out_mt);
  chan_av_kernel<<<dim3(B_ * P_, D_ / 128), 256, 0, stream>>>(abar_c, qzcT, out_mc);
}
